// SparseMultiHeadAttention_75943611728645
// MI455X (gfx1250) — matrix operand order repaired and checked by reading, not compiled, not run on hardware
//
#include <hip/hip_runtime.h>
#include <hip/hip_bf16.h>

typedef _Float16 v16h __attribute__((ext_vector_type(16)));
typedef _Float16 v8h  __attribute__((ext_vector_type(8)));
typedef float    v8f  __attribute__((ext_vector_type(8)));
typedef float    v4f  __attribute__((ext_vector_type(4)));

#define S_LEN 4096
#define D_DIM 1024
#define H_NUM 16
#define DK_DIM 64
#define G_NUM 1
#define KSEL 32
#define NG (S_LEN - 2 * G_NUM)
#define SCALE 0.125f  // 1/sqrt(64)

#if defined(__HIP_DEVICE_COMPILE__) && __has_builtin(__builtin_amdgcn_tensor_load_to_lds) && \
    __has_builtin(__builtin_amdgcn_s_wait_tensorcnt)
#define USE_TDM 1
#else
#define USE_TDM 0
#endif

#if USE_TDM
typedef unsigned int u32;
typedef u32 u32x4 __attribute__((ext_vector_type(4)));
typedef int  i32x4 __attribute__((ext_vector_type(4)));
typedef int  i32x8 __attribute__((ext_vector_type(8)));

// Per-wave B sub-slab in LDS: 64 N-rows x 32 f16 (64B) + 4-DWORD pad = 80B
// pitch (bank-conflict-free, 16B-aligned ds_load_b128 fragment reads).
#define B_ROW_PITCH 80
#define B_WAVE_BYTES (64 * B_ROW_PITCH)  // 5120B per wave per buffer

// TENSOR_LOAD_TO_LDS of a 2D f16 tile: tile_d0=32 (K, contiguous), tile_d1=64
// (N rows), row stride 1024 elements. Descriptor per CDNA5 ISA.
__device__ __forceinline__ void tdm_load_b_slab(u32 lds_byte_off,
                                                const _Float16* gptr) {
  const unsigned long long ga = (unsigned long long)(size_t)gptr;
  u32x4 g0;
  g0.x = 1u;                                   // count=1, is_restore=0
  g0.y = lds_byte_off;                         // lds_addr
  g0.z = (u32)ga;                              // global_addr[31:0]
  g0.w = (u32)((ga >> 32) & 0x01ffffffull)     // global_addr[56:32]
         | (2u << 30);                         // type=2 ("image")
  const u32 td0 = 0x7fffffffu;                 // huge tensor dims -> no OOB clip
  const u32 td1 = 0x7fffffffu;
  const u32 tile0 = 32u, tile1 = 64u;
  const u32 stride0 = 1024u;                   // elements
  // w0: data_size=1 (2B) @ [17:16]; pad_enable @ 20; pad_interval=3 (16 DW)
  //     @ [24:22]; pad_amount=3 (4 DW) @ [31:25]
  const u32 w0 = (1u << 16) | (1u << 20) | (3u << 22) | (3u << 25);
  const u32 w1 = (td0 & 0xffffu) << 16;                     // tensor_dim0[15:0]
  const u32 w2 = (td0 >> 16) | ((td1 & 0xffffu) << 16);     // td0 hi | td1 lo
  const u32 w3 = (td1 >> 16) | (tile0 << 16);               // td1 hi | tile_dim0
  const u32 w4 = tile1 & 0xffffu;                           // tile_dim1, tile_dim2=0
  const u32 w5 = stride0;                                   // dim0 stride [31:0]
  const u32 w6 = 0u;
  const u32 w7 = 0u;
  i32x8 g1 = {(int)w0, (int)w1, (int)w2, (int)w3,
              (int)w4, (int)w5, (int)w6, (int)w7};
  i32x4 gz = {0, 0, 0, 0};
#if __clang_major__ >= 23
  i32x8 gz8 = {0, 0, 0, 0, 0, 0, 0, 0};
  __builtin_amdgcn_tensor_load_to_lds(g0, g1, gz, gz, gz8, 0);
#else
  __builtin_amdgcn_tensor_load_to_lds(g0, g1, gz, gz, 0);
#endif
}
#endif  // USE_TDM

// Pack two 16B halves into a WMMA v16h fragment.
__device__ __forceinline__ v16h pack16(v8h lo, v8h hi) {
  v16h r;
  #pragma unroll
  for (int i = 0; i < 8; ++i) { r[i] = lo[i]; r[i + 8] = hi[i]; }
  return r;
}

// ---------------------------------------------------------------------------
// Weight convert + transpose: WT[n*1024 + k] = (f16) W[k*1024 + n]
// ---------------------------------------------------------------------------
__global__ __launch_bounds__(256) void transpose_cvt_w(
    const float* __restrict__ W, _Float16* __restrict__ WT) {
  __shared__ float tile[32][33];
  const int bx = blockIdx.x * 32;  // n base
  const int by = blockIdx.y * 32;  // k base
  const int tx = threadIdx.x & 31;
  const int ty = threadIdx.x >> 5;
  #pragma unroll
  for (int i = ty; i < 32; i += 8)
    tile[i][tx] = W[(size_t)(by + i) * D_DIM + bx + tx];
  __syncthreads();
  #pragma unroll
  for (int i = ty; i < 32; i += 8)
    WT[(size_t)(bx + i) * D_DIM + by + tx] = (_Float16)tile[tx][i];
}

// ---------------------------------------------------------------------------
// WMMA GEMM: Out[4096 x 1024] = A[4096 x 1024] * W, W transposed (N-major) f16.
// Block = 128 threads (4 waves); workgroup tile 16(M) x 256(N); wave 16 x 64.
// Each wave TDM-streams its own 32x64 B sub-slab into double-buffered private
// LDS. Steady-state loop is branch-free: always issue next slab, then
// s_wait_tensorcnt(1); epilogue iteration waits tensorcnt(0).
// ---------------------------------------------------------------------------
__global__ __launch_bounds__(128) void gemm_f32a_f16out(
    const float* __restrict__ A, const _Float16* __restrict__ WT,
    _Float16* __restrict__ Out) {
  const int lane   = threadIdx.x & 31;
  const int wave   = threadIdx.x >> 5;
  const int tileM  = blockIdx.y * 16;
  const int tileNw = blockIdx.x * 256;
  const int tileN  = tileNw + wave * 64;
  const int row    = tileM + (lane & 15);
  const int kseg   = (lane & 16) ? 16 : 0;
  const int bkseg  = (lane & 16) ? 32 : 0;  // byte offset of K-half in LDS row

  v8f c0 = {}, c1 = {}, c2 = {}, c3 = {};
  const float* arow = A + (size_t)row * D_DIM;

#if USE_TDM
  __shared__ __align__(16) char Bsm[2][4][B_WAVE_BYTES];
  const _Float16* wbase = WT + (size_t)tileN * D_DIM;
#endif

  auto body = [&](int kb, int cur) {
    // A fragment (16x32, fp32 -> f16, contiguous k = 16*(lane/16) + s, as B)
    const float* ap = arow + kb + kseg;
    v4f a0 = *(const v4f*)(ap);
    v4f a1 = *(const v4f*)(ap + 4);
    v4f a2 = *(const v4f*)(ap + 8);
    v4f a3 = *(const v4f*)(ap + 12);
    v16h a;
    #pragma unroll
    for (int i = 0; i < 4; ++i) {
      a[i]      = (_Float16)a0[i];
      a[i + 4]  = (_Float16)a1[i];
      a[i + 8]  = (_Float16)a2[i];
      a[i + 12] = (_Float16)a3[i];
    }
    v16h b0, b1, b2, b3;
#if USE_TDM
    const char* bbase =
        &Bsm[cur][wave][(size_t)(lane & 15) * B_ROW_PITCH + bkseg];
    b0 = pack16(*(const v8h*)(bbase),
                *(const v8h*)(bbase + 16));
    b1 = pack16(*(const v8h*)(bbase + 16 * B_ROW_PITCH),
                *(const v8h*)(bbase + 16 * B_ROW_PITCH + 16));
    b2 = pack16(*(const v8h*)(bbase + 32 * B_ROW_PITCH),
                *(const v8h*)(bbase + 32 * B_ROW_PITCH + 16));
    b3 = pack16(*(const v8h*)(bbase + 48 * B_ROW_PITCH),
                *(const v8h*)(bbase + 48 * B_ROW_PITCH + 16));
#else
    const _Float16* bp =
        WT + (size_t)(tileN + (lane & 15)) * D_DIM + kb + (bkseg >> 1);
    b0 = pack16(*(const v8h*)(bp), *(const v8h*)(bp + 8));
    bp += (size_t)16 * D_DIM;
    b1 = pack16(*(const v8h*)(bp), *(const v8h*)(bp + 8));
    bp += (size_t)16 * D_DIM;
    b2 = pack16(*(const v8h*)(bp), *(const v8h*)(bp + 8));
    bp += (size_t)16 * D_DIM;
    b3 = pack16(*(const v8h*)(bp), *(const v8h*)(bp + 8));
#endif
    c0 = __builtin_amdgcn_wmma_f32_16x16x32_f16(false, a, false, b0, (short)0,
                                                c0, false, false);
    c1 = __builtin_amdgcn_wmma_f32_16x16x32_f16(false, a, false, b1, (short)0,
                                                c1, false, false);
    c2 = __builtin_amdgcn_wmma_f32_16x16x32_f16(false, a, false, b2, (short)0,
                                                c2, false, false);
    c3 = __builtin_amdgcn_wmma_f32_16x16x32_f16(false, a, false, b3, (short)0,
                                                c3, false, false);
  };

#if USE_TDM
  tdm_load_b_slab((u32)(size_t)&Bsm[0][wave][0], wbase);
  int cur = 0;
  for (int kb = 0; kb < D_DIM - 32; kb += 32) {  // steady state: branch-free
    tdm_load_b_slab((u32)(size_t)&Bsm[cur ^ 1][wave][0], wbase + (kb + 32));
    __builtin_amdgcn_s_wait_tensorcnt(1);  // slab kb resident; kb+32 in flight
    __builtin_prefetch(arow + kb + 32 + kseg, 0, 1);
    body(kb, cur);
    cur ^= 1;
  }
  __builtin_amdgcn_s_wait_tensorcnt(0);    // epilogue slab resident
  body(D_DIM - 32, cur);
#else
  for (int kb = 0; kb < D_DIM; kb += 32) {
    if (kb + 32 < D_DIM) __builtin_prefetch(arow + kb + 32 + kseg, 0, 1);
    body(kb, 0);
  }
#endif

  const int colBase = tileN + (lane & 15);
  const int rowBase = tileM + ((lane & 16) ? 8 : 0);
  #pragma unroll
  for (int sub = 0; sub < 4; ++sub) {
    const v8f& c = (sub == 0) ? c0 : (sub == 1) ? c1 : (sub == 2) ? c2 : c3;
    #pragma unroll
    for (int i = 0; i < 8; ++i)
      Out[(size_t)(rowBase + i) * D_DIM + colBase + sub * 16] = (_Float16)c[i];
  }
}

// Variant 2: A is f16 (attention output), output fp32 (final result).
__global__ __launch_bounds__(128) void gemm_f16a_f32out(
    const _Float16* __restrict__ A, const _Float16* __restrict__ WT,
    float* __restrict__ Out) {
  const int lane   = threadIdx.x & 31;
  const int wave   = threadIdx.x >> 5;
  const int tileM  = blockIdx.y * 16;
  const int tileNw = blockIdx.x * 256;
  const int tileN  = tileNw + wave * 64;
  const int row    = tileM + (lane & 15);
  const int kseg   = (lane & 16) ? 16 : 0;
  const int bkseg  = (lane & 16) ? 32 : 0;

  v8f c0 = {}, c1 = {}, c2 = {}, c3 = {};
  const _Float16* arow = A + (size_t)row * D_DIM;

#if USE_TDM
  __shared__ __align__(16) char Bsm[2][4][B_WAVE_BYTES];
  const _Float16* wbase = WT + (size_t)tileN * D_DIM;
#endif

  auto body = [&](int kb, int cur) {
    const _Float16* ap = arow + kb + kseg;
    v16h a = pack16(*(const v8h*)(ap), *(const v8h*)(ap + 8));
    v16h b0, b1, b2, b3;
#if USE_TDM
    const char* bbase =
        &Bsm[cur][wave][(size_t)(lane & 15) * B_ROW_PITCH + bkseg];
    b0 = pack16(*(const v8h*)(bbase),
                *(const v8h*)(bbase + 16));
    b1 = pack16(*(const v8h*)(bbase + 16 * B_ROW_PITCH),
                *(const v8h*)(bbase + 16 * B_ROW_PITCH + 16));
    b2 = pack16(*(const v8h*)(bbase + 32 * B_ROW_PITCH),
                *(const v8h*)(bbase + 32 * B_ROW_PITCH + 16));
    b3 = pack16(*(const v8h*)(bbase + 48 * B_ROW_PITCH),
                *(const v8h*)(bbase + 48 * B_ROW_PITCH + 16));
#else
    const _Float16* bp =
        WT + (size_t)(tileN + (lane & 15)) * D_DIM + kb + (bkseg >> 1);
    b0 = pack16(*(const v8h*)(bp), *(const v8h*)(bp + 8));
    bp += (size_t)16 * D_DIM;
    b1 = pack16(*(const v8h*)(bp), *(const v8h*)(bp + 8));
    bp += (size_t)16 * D_DIM;
    b2 = pack16(*(const v8h*)(bp), *(const v8h*)(bp + 8));
    bp += (size_t)16 * D_DIM;
    b3 = pack16(*(const v8h*)(bp), *(const v8h*)(bp + 8));
#endif
    c0 = __builtin_amdgcn_wmma_f32_16x16x32_f16(false, a, false, b0, (short)0,
                                                c0, false, false);
    c1 = __builtin_amdgcn_wmma_f32_16x16x32_f16(false, a, false, b1, (short)0,
                                                c1, false, false);
    c2 = __builtin_amdgcn_wmma_f32_16x16x32_f16(false, a, false, b2, (short)0,
                                                c2, false, false);
    c3 = __builtin_amdgcn_wmma_f32_16x16x32_f16(false, a, false, b3, (short)0,
                                                c3, false, false);
  };

#if USE_TDM
  tdm_load_b_slab((u32)(size_t)&Bsm[0][wave][0], wbase);
  int cur = 0;
  for (int kb = 0; kb < D_DIM - 32; kb += 32) {
    tdm_load_b_slab((u32)(size_t)&Bsm[cur ^ 1][wave][0], wbase + (kb + 32));
    __builtin_amdgcn_s_wait_tensorcnt(1);
    __builtin_prefetch(arow + kb + 32 + kseg, 0, 1);
    body(kb, cur);
    cur ^= 1;
  }
  __builtin_amdgcn_s_wait_tensorcnt(0);
  body(D_DIM - 32, cur);
#else
  for (int kb = 0; kb < D_DIM; kb += 32) {
    if (kb + 32 < D_DIM) __builtin_prefetch(arow + kb + 32 + kseg, 0, 1);
    body(kb, 0);
  }
#endif

  const int colBase = tileN + (lane & 15);
  const int rowBase = tileM + ((lane & 16) ? 8 : 0);
  #pragma unroll
  for (int sub = 0; sub < 4; ++sub) {
    const v8f& c = (sub == 0) ? c0 : (sub == 1) ? c1 : (sub == 2) ? c2 : c3;
    #pragma unroll
    for (int i = 0; i < 8; ++i)
      Out[(size_t)(rowBase + i) * D_DIM + colBase + sub * 16] = c[i];
  }
}

// ---------------------------------------------------------------------------
// Sparse attention for non-global queries. One wave per (query, head).
// ---------------------------------------------------------------------------
__global__ __launch_bounds__(32) void attn_sel(
    const _Float16* __restrict__ qh, const _Float16* __restrict__ kh,
    const _Float16* __restrict__ vh, const int* __restrict__ idx,
    _Float16* __restrict__ attn) {
  __shared__ float p_s[32];
  __shared__ int   ki_s[32];
  const int j    = threadIdx.x;
  const int nrow = G_NUM + blockIdx.x;
  const int h    = blockIdx.y;
  const int kidx = idx[(size_t)blockIdx.x * KSEL + j];

  const _Float16* qp = qh + (size_t)nrow * D_DIM + h * DK_DIM;
  const _Float16* kp = kh + (size_t)kidx * D_DIM + h * DK_DIM;
  float acc = 0.f;
  #pragma unroll
  for (int c = 0; c < 8; ++c) {
    v8h qv = *(const v8h*)(qp + c * 8);
    v8h kv = *(const v8h*)(kp + c * 8);
    #pragma unroll
    for (int i = 0; i < 8; ++i) acc += (float)qv[i] * (float)kv[i];
  }
  float logit = acc * SCALE;
  float m = logit;
  #pragma unroll
  for (int off = 16; off >= 1; off >>= 1)
    m = fmaxf(m, __shfl_xor(m, off, 32));
  float e = __expf(logit - m);
  float s = e;
  #pragma unroll
  for (int off = 16; off >= 1; off >>= 1) s += __shfl_xor(s, off, 32);
  p_s[j] = e / s;
  ki_s[j] = kidx;
  __syncthreads();
  float o0 = 0.f, o1 = 0.f;
  #pragma unroll 4
  for (int jj = 0; jj < KSEL; ++jj) {
    const _Float16* vp = vh + (size_t)ki_s[jj] * D_DIM + h * DK_DIM;
    const float pj = p_s[jj];
    o0 += pj * (float)vp[j];
    o1 += pj * (float)vp[j + 32];
  }
  _Float16* op = attn + (size_t)nrow * D_DIM + h * DK_DIM;
  op[j]      = (_Float16)o0;
  op[j + 32] = (_Float16)o1;
}

// ---------------------------------------------------------------------------
// Dense attention for the 2 global queries (rows 0 and S-1).
// ---------------------------------------------------------------------------
__global__ __launch_bounds__(256) void attn_global(
    const _Float16* __restrict__ qh, const _Float16* __restrict__ kh,
    const _Float16* __restrict__ vh, _Float16* __restrict__ attn) {
  __shared__ float logits[S_LEN];
  __shared__ float q_s[DK_DIM];
  __shared__ float red[256];
  const int t    = threadIdx.x;
  const int h    = blockIdx.y;
  const int nrow = (blockIdx.x == 0) ? 0 : (S_LEN - 1);

  if (t < DK_DIM) q_s[t] = (float)qh[(size_t)nrow * D_DIM + h * DK_DIM + t];
  __syncthreads();

  float lmax = -3.4e38f;
  for (int key = t; key < S_LEN; key += 256) {
    const _Float16* kp = kh + (size_t)key * D_DIM + h * DK_DIM;
    float acc = 0.f;
    #pragma unroll
    for (int d = 0; d < DK_DIM; ++d) acc += q_s[d] * (float)kp[d];
    const float lg = acc * SCALE;
    logits[key] = lg;
    lmax = fmaxf(lmax, lg);
  }
  red[t] = lmax;
  __syncthreads();
  for (int o = 128; o >= 1; o >>= 1) {
    if (t < o) red[t] = fmaxf(red[t], red[t + o]);
    __syncthreads();
  }
  const float m = red[0];
  __syncthreads();
  float lsum = 0.f;
  for (int key = t; key < S_LEN; key += 256) {
    const float e = __expf(logits[key] - m);
    logits[key] = e;
    lsum += e;
  }
  red[t] = lsum;
  __syncthreads();
  for (int o = 128; o >= 1; o >>= 1) {
    if (t < o) red[t] += red[t + o];
    __syncthreads();
  }
  const float inv = 1.f / red[0];
  __syncthreads();
  if (t < DK_DIM) {
    float acc = 0.f;
    for (int key = 0; key < S_LEN; ++key)
      acc += logits[key] * (float)vh[(size_t)key * D_DIM + h * DK_DIM + t];
    attn[(size_t)nrow * D_DIM + h * DK_DIM + t] = (_Float16)(acc * inv);
  }
}

// ---------------------------------------------------------------------------
// Host orchestration. Workspace (f16 elems): 4x1M weightsT + 3x4M qkv + 4M attn.
// ---------------------------------------------------------------------------
extern "C" void kernel_launch(void* const* d_in, const int* in_sizes, int n_in,
                              void* d_out, int out_size, void* d_ws,
                              size_t ws_size, hipStream_t stream) {
  const float* Q  = (const float*)d_in[0];
  const float* K  = (const float*)d_in[1];
  const float* V  = (const float*)d_in[2];
  const float* Wq = (const float*)d_in[3];
  const float* Wk = (const float*)d_in[4];
  const float* Wv = (const float*)d_in[5];
  const float* Wo = (const float*)d_in[6];
  const int*   idx = (const int*)d_in[7];
  float* out = (float*)d_out;

  _Float16* p = (_Float16*)d_ws;
  const size_t WSZ = (size_t)D_DIM * D_DIM;
  const size_t ASZ = (size_t)S_LEN * D_DIM;
  _Float16* wqT = p;            p += WSZ;
  _Float16* wkT = p;            p += WSZ;
  _Float16* wvT = p;            p += WSZ;
  _Float16* woT = p;            p += WSZ;
  _Float16* q_h = p;            p += ASZ;
  _Float16* k_h = p;            p += ASZ;
  _Float16* v_h = p;            p += ASZ;
  _Float16* att = p;            p += ASZ;

  {
    dim3 g(D_DIM / 32, D_DIM / 32), b(256);
    transpose_cvt_w<<<g, b, 0, stream>>>(Wq, wqT);
    transpose_cvt_w<<<g, b, 0, stream>>>(Wk, wkT);
    transpose_cvt_w<<<g, b, 0, stream>>>(Wv, wvT);
    transpose_cvt_w<<<g, b, 0, stream>>>(Wo, woT);
  }
  {
    dim3 g(D_DIM / 256, S_LEN / 16), b(128);
    gemm_f32a_f16out<<<g, b, 0, stream>>>(Q, wqT, q_h);
    gemm_f32a_f16out<<<g, b, 0, stream>>>(K, wkT, k_h);
    gemm_f32a_f16out<<<g, b, 0, stream>>>(V, wvT, v_h);
  }
  attn_sel<<<dim3(NG, H_NUM), 32, 0, stream>>>(q_h, k_h, v_h, idx, att);
  attn_global<<<dim3(2, H_NUM), 256, 0, stream>>>(q_h, k_h, v_h, att);
  {
    dim3 g(D_DIM / 256, S_LEN / 16), b(128);
    gemm_f16a_f32out<<<g, b, 0, stream>>>(att, woT, out);
  }
}
